// SeqAttention_54975581389358
// MI455X (gfx1250) — compile-verified, matrix operand order checked by reading, not run on hardware
//
#include <hip/hip_runtime.h>
#include <hip/hip_bf16.h>
#include <math.h>

// Problem constants (from reference)
#define BB 4
#define HB 8          // heads
#define BHN 32        // B*H
#define MM 1024       // queries
#define LLN 3200      // kv length
#define DD 128        // head dim
#define HIDN 1024
#define RAMPN 128
#define SPAN_SIZE 2048.0f
#define NT 50         // LLN / 64 kv tiles
#define MT 128        // query rows per workgroup
#define CAUSAL_K 2175 // max_mem - M - 1
#define INV_SQRT_D 0.08838834764831844f

typedef __attribute__((ext_vector_type(16))) __bf16 v16bf;
typedef __attribute__((ext_vector_type(8)))  __bf16 v8bf;
typedef __attribute__((ext_vector_type(4)))  __bf16 v4bf;
typedef __attribute__((ext_vector_type(8)))  float  v8f;

// QP stored transposed: qp_t[r][row_local], padded row stride
//  - writes: one lane's 8 C-fragment rows are contiguous -> single ds_store_b128
//  - reads:  consecutive lanes differ by 1 in r -> 272B apart -> 4 banks apart (conflict-free)
#define QP_STRIDE 136                               // elements (272 B, 16B-aligned rows)

// LDS layout (bytes) -- exactly 320 KB
#define QP_BYTES   ((size_t)MM * QP_STRIDE * 2)     // 278528  bf16 [1024][136]
#define PBUF_OFF   QP_BYTES                         // 16 KB   8 waves x [16][64] bf16
#define STAGE_OFF  (QP_BYTES + 8 * 16 * 64 * 2)     // 32 KB   K[64][128] + V^T[128][64] bf16
#define SMEM_BYTES (STAGE_OFF + (64 * 128 + 128 * 64) * 2)   // 327680 = 320 KB

// ---- WMMA fragment helpers (CDNA5 16x16x32 bf16 layouts, wave32) -----------
// A 16x32 bf16: lane L -> row L%16; element j -> k = (j&7) + ((j>>3)<<4) + 8*(L/16)
__device__ __forceinline__ v8f wmma_bf16(v16bf a, v16bf b, v8f c) {
  return __builtin_amdgcn_wmma_f32_16x16x32_bf16(false, a, false, b, (short)0, c, false, false);
}

__device__ __forceinline__ void vzero(v8f& v) {
#pragma unroll
  for (int j = 0; j < 8; ++j) v[j] = 0.f;
}

// fragment from bf16 LDS, row-major pseudo-A (row stride = stride elems): 2x ds_load_b128
__device__ __forceinline__ v16bf frag_lds(const __bf16* base, int stride, int lane) {
  const __bf16* p = base + (size_t)(lane & 15) * stride + ((lane >> 4) << 3);
  v8bf lo = *(const v8bf*)p;          // k = 0..7   (+8*half)
  v8bf hi = *(const v8bf*)(p + 16);   // k = 16..23 (+8*half)
  return __builtin_shufflevector(lo, hi, 0, 1, 2, 3, 4, 5, 6, 7, 8, 9, 10, 11, 12, 13, 14, 15);
}

// fragment from fp32 global, row-major pseudo-A: 4x global_load_b128 + cvt
__device__ __forceinline__ v16bf frag_global(const float* __restrict__ base, int ld, int lane) {
  const float* rp = base + (size_t)(lane & 15) * ld + ((lane >> 4) << 3);
  float4 a = *(const float4*)(rp);
  float4 b = *(const float4*)(rp + 4);
  float4 c = *(const float4*)(rp + 16);
  float4 d = *(const float4*)(rp + 20);
  v16bf f;
  f[0] = (__bf16)a.x; f[1] = (__bf16)a.y; f[2]  = (__bf16)a.z; f[3]  = (__bf16)a.w;
  f[4] = (__bf16)b.x; f[5] = (__bf16)b.y; f[6]  = (__bf16)b.z; f[7]  = (__bf16)b.w;
  f[8] = (__bf16)c.x; f[9] = (__bf16)c.y; f[10] = (__bf16)c.z; f[11] = (__bf16)c.w;
  f[12] = (__bf16)d.x; f[13] = (__bf16)d.y; f[14] = (__bf16)d.z; f[15] = (__bf16)d.w;
  return f;
}

// ---- Kernel 1: expire-span (remaining_offset, per-tile max, aux loss) -------
__global__ __launch_bounds__(256) void span_kernel(
    const float* __restrict__ hid, const float* __restrict__ cnt,
    const float* __restrict__ sw, const float* __restrict__ sb,
    float* __restrict__ ro_out, float* __restrict__ aux, float* __restrict__ tile_max)
{
  int b = blockIdx.x / NT, t = blockIdx.x % NT;
  int wid = threadIdx.x >> 5, lane = threadIdx.x & 31;
  __shared__ float smax[8], saux[8];
  float wmax = -INFINITY, waux = 0.f;
  float bias = sb[0];
  for (int i = 0; i < 8; ++i) {
    int l = t * 64 + wid * 8 + i;
    const float* hp = hid + ((size_t)b * LLN + l) * HIDN;
    float s = 0.f;
#pragma unroll 4
    for (int j = 0; j < HIDN / 32; ++j) s += hp[j * 32 + lane] * sw[j * 32 + lane];
#pragma unroll
    for (int off = 16; off; off >>= 1) s += __shfl_xor(s, off, 32);
    float mspan = SPAN_SIZE / (1.f + __expf(-(s * (1.f / 16.f) + bias)));
    float ro = mspan - cnt[(size_t)b * LLN + l];
    if (lane == 0) {
      ro_out[(size_t)b * LLN + l] = ro;
      wmax = fmaxf(wmax, ro);
      // aux ramp region: m in (ro, ro+128) intersect [0, M-1]; sum of (ro - m)
      int mlo = (int)floorf(ro) + 1; if (mlo < 0) mlo = 0;
      int mhi = (int)ceilf(ro + (float)RAMPN) - 1; if (mhi > MM - 1) mhi = MM - 1;
      if (mhi >= mlo) {
        float nn = (float)(mhi - mlo + 1);
        waux += nn * ro - 0.5f * nn * (float)(mlo + mhi);
      }
    }
  }
  if (lane == 0) { smax[wid] = wmax; saux[wid] = waux; }
  __syncthreads();
  if (threadIdx.x == 0) {
    float m = -INFINITY, a = 0.f;
    for (int w = 0; w < 8; ++w) { m = fmaxf(m, smax[w]); a += saux[w]; }
    tile_max[b * NT + t] = m;
    atomicAdd(&aux[b], a * (1e-6f / ((float)RAMPN * (float)MM)));
  }
}

// ---- Kernel 2: fused flash attention with expire-span mask + rel-pos bias ---
__global__ __launch_bounds__(256) void attn_kernel(
    const float* __restrict__ Q, const float* __restrict__ Kv, const float* __restrict__ Vv,
    const float* __restrict__ PE, const float* __restrict__ RO,
    const float* __restrict__ tile_max, float* __restrict__ Out)
{
  extern __shared__ char smem[];
  __bf16* qp    = (__bf16*)smem;                       // [1024][QP_STRIDE] transposed
  __bf16* pbufA = (__bf16*)(smem + PBUF_OFF);          // 8 x [16][64]
  __bf16* kst   = (__bf16*)(smem + STAGE_OFF);         // [64][128]   (K tile / PE chunk)
  __bf16* vst   = kst + 64 * 128;                      // [128][64]   (V^T tile)

  int bh = blockIdx.y, b = bh >> 3;
  int m0 = blockIdx.x * MT;
  int tid = threadIdx.x;
  int wid = tid >> 5, lane = tid & 31;
  int hh = lane >> 4, nn = lane & 15;
  int ms = m0 + wid * 16;                              // this wave's row strip
  __bf16* pbuf = pbufA + wid * (16 * 64);

  // --- Q strip fragments (16x128 as 4 bf16 A-fragments), b128 global loads ---
  v16bf qa[4];
  const float* qbase = Q + ((size_t)bh * MM + ms) * DD;
#pragma unroll
  for (int kc = 0; kc < 4; ++kc) qa[kc] = frag_global(qbase + kc * 32, DD, lane);

  // --- QP = Q_strip @ key_pe  (16 x 1024) -> LDS bf16, stored TRANSPOSED ---
  // stage key_pe 128x64 chunks transposed through kst so fragments are contiguous
  for (int c0 = 0; c0 < MM; c0 += 64) {
    __syncthreads();
#pragma unroll
    for (int i = 0; i < 8; ++i) {
      int f = tid + 256 * i;            // 128*64/4 = 2048 float4s
      int d = f >> 4;                   // PE row (head-dim)
      int r0 = (f & 15) * 4;            // rel-pos column within chunk
      float4 p4 = *(const float4*)(PE + (size_t)d * MM + c0 + r0);
      kst[(size_t)(r0 + 0) * 128 + d] = (__bf16)p4.x;
      kst[(size_t)(r0 + 1) * 128 + d] = (__bf16)p4.y;
      kst[(size_t)(r0 + 2) * 128 + d] = (__bf16)p4.z;
      kst[(size_t)(r0 + 3) * 128 + d] = (__bf16)p4.w;
    }
    __syncthreads();
#pragma unroll
    for (int cs = 0; cs < 4; ++cs) {
      v8f acc; vzero(acc);
#pragma unroll
      for (int kc = 0; kc < 4; ++kc)
        acc = wmma_bf16(qa[kc], frag_lds(kst + (size_t)(cs * 16) * 128 + kc * 32, 128, lane), acc);
      // one lane's 8 rows are contiguous in qp_t[r][.] -> single ds_store_b128
      v8bf pk;
#pragma unroll
      for (int v = 0; v < 8; ++v) pk[v] = (__bf16)acc[v];
      *(v8bf*)(qp + (size_t)(c0 + cs * 16 + nn) * QP_STRIDE + wid * 16 + 8 * hh) = pk;
    }
  }

  // --- online softmax state (rows replicated across each 16-lane half) ---
  v8f oacc[8];
#pragma unroll
  for (int d = 0; d < 8; ++d) vzero(oacc[d]);
  float mrow[8], s0[8], s1[8];
#pragma unroll
  for (int v = 0; v < 8; ++v) { mrow[v] = -INFINITY; s0[v] = 0.f; s1[v] = 0.f; }

  int t_end = (m0 + MT - 1 + CAUSAL_K) / 64 + 1;
  if (t_end > NT) t_end = NT;
  const float skip_lim = (float)m0 - (float)RAMPN;

  for (int t = 0; t < t_end; ++t) {
    if (tile_max[b * NT + t] <= skip_lim) continue;   // whole tile emask == 0 (uniform)
    int l0 = t * 64;
    if (t + 1 < t_end) {                              // global_prefetch_b8 next tiles
      __builtin_prefetch(Kv + ((size_t)bh * LLN + l0 + 64) * DD, 0, 1);
      __builtin_prefetch(Vv + ((size_t)bh * LLN + l0 + 64) * DD, 0, 1);
    }

    // --- cooperative bf16 staging: K row-major, V transposed ---
    __syncthreads();                                  // previous tile's frag reads done
    {
      const float* kbase = Kv + ((size_t)bh * LLN + l0) * DD;
      const float* vbase = Vv + ((size_t)bh * LLN + l0) * DD;
#pragma unroll
      for (int i = 0; i < 8; ++i) {
        int f = tid + 256 * i;          // 64*128/4 = 2048 float4s
        int l = f >> 5;
        int d0 = (f & 31) * 4;
        float4 k4 = *(const float4*)(kbase + (size_t)l * DD + d0);
        v4bf kb; kb[0] = (__bf16)k4.x; kb[1] = (__bf16)k4.y; kb[2] = (__bf16)k4.z; kb[3] = (__bf16)k4.w;
        *(v4bf*)(kst + (size_t)l * 128 + d0) = kb;    // ds_store_b64
        float4 v4 = *(const float4*)(vbase + (size_t)l * DD + d0);
        vst[(size_t)(d0 + 0) * 64 + l] = (__bf16)v4.x;
        vst[(size_t)(d0 + 1) * 64 + l] = (__bf16)v4.y;
        vst[(size_t)(d0 + 2) * 64 + l] = (__bf16)v4.z;
        vst[(size_t)(d0 + 3) * 64 + l] = (__bf16)v4.w;
      }
    }
    __syncthreads();

    // --- S = Q_strip @ K_tile^T  (16 x 64), fragments from LDS ---
    v8f S[4];
#pragma unroll
    for (int sub = 0; sub < 4; ++sub) {
      v8f s; vzero(s);
#pragma unroll
      for (int kc = 0; kc < 4; ++kc)
        s = wmma_bf16(qa[kc], frag_lds(kst + (size_t)(sub * 16) * 128 + kc * 32, 128, lane), s);
      S[sub] = s;
    }

    float roc[4];
#pragma unroll
    for (int sub = 0; sub < 4; ++sub) roc[sub] = RO[(size_t)b * LLN + l0 + sub * 16 + nn];

    // --- bias + scale + mask; tile row-max ---
    float tmax[8];
#pragma unroll
    for (int v = 0; v < 8; ++v) tmax[v] = -INFINITY;
#pragma unroll
    for (int sub = 0; sub < 4; ++sub) {
#pragma unroll
      for (int v = 0; v < 8; ++v) {
        int rowl = wid * 16 + v + 8 * hh;              // tile-local row
        int row = m0 + rowl;                           // global query index
        int l = l0 + sub * 16 + nn;
        float s = S[sub][v];
        int r = l - row - 1152;                        // rel-pos bias (skew)
        if ((unsigned)r < 1024u)
          s += (float)qp[(size_t)r * QP_STRIDE + rowl];
        s *= INV_SQRT_D;
        float em = fminf(fmaxf((roc[sub] - (float)row) * (1.f / (float)RAMPN) + 1.f, 0.f), 1.f);
        if (l > row + CAUSAL_K || em <= 0.f) s = -INFINITY;
        S[sub][v] = s;
        tmax[v] = fmaxf(tmax[v], s);
      }
    }
#pragma unroll
    for (int v = 0; v < 8; ++v) {                      // reduce within 16-lane half
      float x = tmax[v];
      x = fmaxf(x, __shfl_xor(x, 1, 32));
      x = fmaxf(x, __shfl_xor(x, 2, 32));
      x = fmaxf(x, __shfl_xor(x, 4, 32));
      x = fmaxf(x, __shfl_xor(x, 8, 32));
      tmax[v] = x;
    }

    float scl[8], ps0[8], ps1[8];
#pragma unroll
    for (int v = 0; v < 8; ++v) {
      float mn = fmaxf(mrow[v], tmax[v]);
      scl[v] = (mrow[v] == -INFINITY) ? 0.f : __expf(mrow[v] - mn);
      mrow[v] = mn; ps0[v] = 0.f; ps1[v] = 0.f;
    }
#pragma unroll
    for (int sub = 0; sub < 4; ++sub) {
#pragma unroll
      for (int v = 0; v < 8; ++v) {
        int row = ms + v + 8 * hh;
        float s = S[sub][v];
        float p = (s == -INFINITY) ? 0.f : __expf(s - mrow[v]);
        float em = fminf(fmaxf((roc[sub] - (float)row) * (1.f / (float)RAMPN) + 1.f, 0.f), 1.f);
        float pm = p * em;
        ps0[v] += p; ps1[v] += pm;
        pbuf[(size_t)(v + 8 * hh) * 64 + sub * 16 + nn] = (__bf16)pm;   // transpose via LDS
      }
    }
#pragma unroll
    for (int v = 0; v < 8; ++v) {
      float a = ps0[v], c = ps1[v];
      a += __shfl_xor(a, 1, 32); c += __shfl_xor(c, 1, 32);
      a += __shfl_xor(a, 2, 32); c += __shfl_xor(c, 2, 32);
      a += __shfl_xor(a, 4, 32); c += __shfl_xor(c, 4, 32);
      a += __shfl_xor(a, 8, 32); c += __shfl_xor(c, 8, 32);
      s0[v] = s0[v] * scl[v] + a;
      s1[v] = s1[v] * scl[v] + c;
    }
#pragma unroll
    for (int d = 0; d < 8; ++d)
#pragma unroll
      for (int v = 0; v < 8; ++v) oacc[d][v] *= scl[v];

    // --- O += P @ V_tile ; P from per-wave LDS, V^T fragments contiguous ---
    v16bf pa[2];
#pragma unroll
    for (int kc2 = 0; kc2 < 2; ++kc2)
      pa[kc2] = frag_lds(pbuf + kc2 * 32, 64, lane);
#pragma unroll
    for (int dsub = 0; dsub < 8; ++dsub) {
#pragma unroll
      for (int kc2 = 0; kc2 < 2; ++kc2) {
        v16bf vb = frag_lds(vst + (size_t)(dsub * 16) * 64 + kc2 * 32, 64, lane);
        oacc[dsub] = wmma_bf16(pa[kc2], vb, oacc[dsub]);
      }
    }
  }

  // out = acc / (S1 + 1e-8 * S0)   (mask-renormalized softmax, same max shift)
#pragma unroll
  for (int v = 0; v < 8; ++v) {
    float inv = 1.f / (s1[v] + 1e-8f * s0[v]);
#pragma unroll
    for (int dsub = 0; dsub < 8; ++dsub)
      Out[((size_t)bh * MM + ms + v + 8 * hh) * DD + dsub * 16 + nn] = oacc[dsub][v] * inv;
  }
}

// ---- launch -----------------------------------------------------------------
extern "C" void kernel_launch(void* const* d_in, const int* in_sizes, int n_in,
                              void* d_out, int out_size, void* d_ws, size_t ws_size,
                              hipStream_t stream) {
  const float* Q   = (const float*)d_in[0];
  const float* Kp  = (const float*)d_in[1];
  const float* Vp  = (const float*)d_in[2];
  const float* HID = (const float*)d_in[3];
  const float* CNT = (const float*)d_in[4];
  const float* PE  = (const float*)d_in[5];
  const float* SW  = (const float*)d_in[6];
  const float* SB  = (const float*)d_in[7];

  float* out = (float*)d_out;                       // (BH, M, D)
  float* aux = out + (size_t)BHN * MM * DD;         // (B,)
  float* ro  = aux + BB;                            // (B, L)
  float* tmax = (float*)d_ws;                       // (B, NT) per-tile max remaining_offset

  hipMemsetAsync(aux, 0, BB * sizeof(float), stream);
  span_kernel<<<dim3(BB * NT), dim3(256), 0, stream>>>(HID, CNT, SW, SB, ro, aux, tmax);

  attn_kernel<<<dim3(MM / MT, BHN), dim3(256), SMEM_BYTES, stream>>>(Q, Kp, Vp, PE, ro, tmax, out);
}
